// GAN_61280593379380
// MI455X (gfx1250) — matrix operand order repaired and checked by reading, not compiled, not run on hardware
//
#include <hip/hip_runtime.h>
#include <hip/hip_bf16.h>

// ---------------------------------------------------------------------------
// Problem constants (fixed by the reference)
// ---------------------------------------------------------------------------
#define V 32000   // vocab
#define H 5       // hidden
#define T 1024    // encoder steps
#define L 1024    // decoder steps

typedef _Float16 v16h __attribute__((ext_vector_type(16)));
typedef float    v8f  __attribute__((ext_vector_type(8)));
typedef unsigned int uint32x4 __attribute__((ext_vector_type(4)));
typedef int          int32x4  __attribute__((ext_vector_type(4)));
typedef int          int32x8  __attribute__((ext_vector_type(8)));

// k1 tiling: 64 row-tiles of 16 rows x 40 K-chunks of 800 => 2560 waves,
// each doing 800/32 = 25 v_wmma_f32_16x16x32_f16 MACs.
constexpr int KCH    = 40;
constexpr int KLEN   = V / KCH;     // 800
constexpr int KSTEPS = KLEN / 32;   // 25
constexpr int TILES  = T / 16;      // 64

// decoder persistent grid: 100 WGs x 320 threads = 32000 lanes (1 vocab/lane)
constexpr int NWG = 100;
constexpr int TPB = 320;            // 10 waves (wave32)

// ---------------------------------------------------------------------------
// k0: re-zero sync state every launch (graph-replay safe, deterministic)
// ---------------------------------------------------------------------------
__global__ void k0_init(int* __restrict__ ready, int* __restrict__ cnt) {
  int i = blockIdx.x * blockDim.x + threadIdx.x;
  if (i < L) { ready[i] = 0; cnt[i] = 0; }
}

// ---------------------------------------------------------------------------
// k1: xe_all[T,H] partials = x @ Eg_en^T via WMMA (f16 in, f32 accumulate).
// One wave per (16-row tile, 800-wide K chunk). The B panel (5 rows x 800
// floats of Eg_en) is staged into LDS once per block by the Tensor Data
// Mover (tensor_load_to_lds + s_wait_tensorcnt), then read branchlessly.
// The A side streams x (131 MB, the HBM-bound part) with clause-merged
// global_load_b128 + global_prefetch_b8.
// ---------------------------------------------------------------------------
__global__ __launch_bounds__(32) void k1_xe_wmma(const float* __restrict__ x,
                                                 const float* __restrict__ Eg_en,
                                                 float* __restrict__ part1) {
  __shared__ float Bpanel[H * KLEN];   // 16000 B

  const int tile  = blockIdx.x / KCH;
  const int kc    = blockIdx.x % KCH;
  const int lane  = threadIdx.x;      // 0..31
  const int hi    = lane >> 4;        // half-wave select
  const int lo    = lane & 15;        // A: row m ; B/D: col n
  const int t0    = tile * 16;
  const int kbase = kc * KLEN;

  // ---- TDM: stage Eg_en[:, kbase : kbase+KLEN] (5 x 800 f32) into LDS ----
  {
    const unsigned long long gaddr =
        (unsigned long long)(const void*)(Eg_en + kbase);
    // Low 32 bits of a flat LDS address are the workgroup-relative offset.
    const unsigned ldsoff = (unsigned)(unsigned long long)(const void*)Bpanel;

    // D# group 0 (ISA 8.3): count=1 | lds_addr | global_addr[56:0] | type=2
    uint32x4 g0;
    g0[0] = 1u;                                    // count=1, user mode
    g0[1] = ldsoff;                                // lds_addr[31:0]
    g0[2] = (unsigned)(gaddr & 0xffffffffu);       // global_addr[31:0]
    g0[3] = (unsigned)((gaddr >> 32) & 0x01ffffffu) | 0x80000000u; // type=2

    // D# group 1 (ISA 8.4): data_size=4B; tensor_dim0=V; tensor_dim1=H;
    // tile_dim0=KLEN; tile_dim1=H; tensor_dim0_stride=V
    int32x8 g1;
    g1[0] = 0x00020000;                 // data_size=2 (4 bytes), mask=0
    g1[1] = (int)((V & 0xffff) << 16);  // tensor_dim0[15:0] -> bits[63:48]
    g1[2] = (int)(((V >> 16) & 0xffff)  // tensor_dim0[31:16]
                  | ((H & 0xffff) << 16));          // tensor_dim1[15:0]
    g1[3] = (int)(((H >> 16) & 0xffff)              // tensor_dim1[31:16]
                  | ((KLEN & 0xffff) << 16));       // tile_dim0
    g1[4] = (int)(H & 0xffff);                      // tile_dim1 (tile_dim2=0)
    g1[5] = (int)V;                                 // tensor_dim0_stride[31:0]
    g1[6] = 0;                                      // stride[47:32], dim1_stride
    g1[7] = 0;

    int32x4 z4 = {0, 0, 0, 0};
#if defined(__clang_major__) && (__clang_major__ >= 23)
    int32x8 z8 = {0, 0, 0, 0, 0, 0, 0, 0};
    __builtin_amdgcn_tensor_load_to_lds(g0, g1, z4, z4, z8, 0);
#else
    __builtin_amdgcn_tensor_load_to_lds(g0, g1, z4, z4, 0);
#endif
    __builtin_amdgcn_s_wait_tensorcnt(0);
  }

  // Branchless B row select: clamp row, zero via multiply (no exec waves).
  const int   brow  = (lo < H) ? lo : 0;
  const float bmask = (lo < H) ? 1.f : 0.f;
  const float* __restrict__ brp = &Bpanel[brow * KLEN];

  v8f acc = {};
  for (int s = 0; s < KSTEPS; ++s) {
    const int v0 = kbase + s * 32;

    // Stream-prefetch ahead on the A (x) stream (global_prefetch_b8).
    __builtin_prefetch(&x[(size_t)(t0 + lo) * V + v0 + 32], 0, 0);

    // A fragment: 16x32 f16 (ISA 7.12.2 16-bit A layout):
    // VGPR j in 0..3 -> K = 2j + 8*hi ; j in 4..7 -> K = 16+2(j-4) + 8*hi
    v16h a;
#pragma unroll
    for (int j = 0; j < 8; ++j) {
      const int k0 = ((j < 4) ? (2 * j) : (16 + 2 * (j - 4))) + 8 * hi;
      const float* px = &x[(size_t)(t0 + lo) * V + v0 + k0];
      a[2 * j]     = (_Float16)px[0];
      a[2 * j + 1] = (_Float16)px[1];
    }

    // B fragment from LDS: B[k][n] = Eg_en[n, v0+k] (n<H else 0):
    // same K order as A: VGPR j in 0..3 -> K = 2j + 8*hi ; j in 4..7 -> K = 16+2(j-4) + 8*hi
    v16h b;
#pragma unroll
    for (int j = 0; j < 8; ++j) {
      const int k = ((j < 4) ? (2 * j) : (16 + 2 * (j - 4))) + 8 * hi + (s * 32);
      b[2 * j]     = (_Float16)(brp[k]     * bmask);
      b[2 * j + 1] = (_Float16)(brp[k + 1] * bmask);
    }

    acc = __builtin_amdgcn_wmma_f32_16x16x32_f16(
        /*neg_a=*/false, a, /*neg_b=*/false, b,
        /*c_mod=*/(short)0, acc, /*reuse_a=*/false, /*reuse_b=*/false);
  }

  // D layout: lane lo = n, VGPR r -> row m = r + 8*hi. Keep n < H only.
  if (lo < H) {
#pragma unroll
    for (int r = 0; r < 8; ++r) {
      const int row = t0 + r + 8 * hi;
      part1[(size_t)kc * (T * H) + row * H + lo] = acc[r];
    }
  }
}

// Deterministic fixed-order reduction over the 40 K-chunk partials.
__global__ void k1b_reduce(const float* __restrict__ part1,
                           float* __restrict__ xe_all) {
  int i = blockIdx.x * blockDim.x + threadIdx.x;
  if (i < T * H) {
    float s = 0.f;
#pragma unroll 8
    for (int kc = 0; kc < KCH; ++kc) s += part1[(size_t)kc * (T * H) + i];
    xe_all[i] = s;
  }
}

// ---------------------------------------------------------------------------
// k2: serial encoder GRU scan (1024 steps). One block; gates evaluated by
// 10 lanes (z, r), then 5 lanes (candidate); weights/state in LDS. Writes s_last -> sbuf[0],
// then releases ready[0] for the decoder.
// ---------------------------------------------------------------------------
__global__ __launch_bounds__(128) void k2_encoder(const float* __restrict__ xe_all,
                                                  const float* __restrict__ Wg,
                                                  const float* __restrict__ Ug,
                                                  const float* __restrict__ bg,
                                                  float* __restrict__ sbuf,
                                                  int* __restrict__ ready) {
  __shared__ float W[3 * H * H], U[3 * H * H], Bv[3 * H];
  __shared__ float s[H], snew[H], zr[2 * H], xe[H];
  const int tid = threadIdx.x;

  if (tid < 3 * H * H) { W[tid] = Wg[tid]; U[tid] = Ug[tid]; }
  if (tid < 3 * H)     Bv[tid] = bg[tid];
  if (tid < H)         s[tid] = 0.f;
  __syncthreads();

  for (int t = 0; t < T; ++t) {
    if (tid < H) xe[tid] = xe_all[t * H + tid];
    __syncthreads();
    // phase A: z (lanes 0..4) and r (lanes 5..9)
    if (tid < 2 * H) {
      const int g = tid / H, i = tid % H;
      float p = Bv[g * H + i];
#pragma unroll
      for (int j = 0; j < H; ++j)
        p += W[(g * H + i) * H + j] * xe[j] + U[(g * H + i) * H + j] * s[j];
      zr[tid] = 1.f / (1.f + __expf(-p));
    }
    __syncthreads();
    // phase B: candidate + blend (lanes 0..4)
    if (tid < H) {
      float p = Bv[2 * H + tid];
#pragma unroll
      for (int j = 0; j < H; ++j)
        p += W[(2 * H + tid) * H + j] * xe[j] +
             U[(2 * H + tid) * H + j] * (s[j] * zr[H + j]);
      const float c = tanhf(p);
      snew[tid] = zr[tid] * s[tid] + (1.f - zr[tid]) * c;
    }
    __syncthreads();
    if (tid < H) s[tid] = snew[tid];
    __syncthreads();
  }

  if (tid < H) sbuf[tid] = s[tid];
  __threadfence();
  if (tid == 0)
    __hip_atomic_store(&ready[0], 1, __ATOMIC_RELEASE, __HIP_MEMORY_SCOPE_AGENT);
}

// ---------------------------------------------------------------------------
// k3: persistent decoder. out[t] = relu(Eg_de @ s_t + c);
//     s_{t+1} = GRU(s_t, Eg_en @ out[t]). One vocab element per lane
//     (registers); one global sync per step; last-arriving WG does the
//     cross-WG xe sum + 5x5 GRU update and releases ready[t+1].
// ---------------------------------------------------------------------------
__global__ __launch_bounds__(TPB) void k3_decoder(const float* __restrict__ Eg_en,
                                                  const float* __restrict__ Eg_de,
                                                  const float* __restrict__ cg_de,
                                                  const float* __restrict__ Wg,
                                                  const float* __restrict__ Ug,
                                                  const float* __restrict__ bg,
                                                  float* __restrict__ out,
                                                  float* __restrict__ sbuf,
                                                  float* __restrict__ part,
                                                  int* __restrict__ ready,
                                                  int* __restrict__ cnt) {
  __shared__ float W[3 * H * H], U[3 * H * H], Bv[3 * H];
  __shared__ float scur[H], red[H], zr[2 * H];
  __shared__ int islast;

  const int tid  = threadIdx.x;
  const int wg   = blockIdx.x;
  const int v    = wg * TPB + tid;           // this lane's vocab element
  const int lane = tid & 31;

  if (tid < 3 * H * H) { W[tid] = Wg[tid]; U[tid] = Ug[tid]; }
  if (tid < 3 * H)     Bv[tid] = bg[tid];

  float ed[H], ee[H];
#pragma unroll
  for (int h = 0; h < H; ++h) {
    ed[h] = Eg_de[(size_t)v * H + h];        // row of Eg_de  [V,H]
    ee[h] = Eg_en[(size_t)h * V + v];        // column of Eg_en [H,V]
  }
  const float cgv = cg_de[v];
  __syncthreads();

  for (int t = 0; t < L; ++t) {
    // --- wait for s_t ---
    if (tid == 0) {
      while (__hip_atomic_load(&ready[t], __ATOMIC_ACQUIRE,
                               __HIP_MEMORY_SCOPE_AGENT) == 0)
        __builtin_amdgcn_s_sleep(1);
    }
    __syncthreads();
    if (tid < H) scur[tid] = sbuf[t * H + tid];
    if (tid < H) red[tid] = 0.f;
    if (tid == 0) islast = 0;
    __syncthreads();

    // --- w_t for this lane's vocab element; write output row ---
    float d = cgv;
#pragma unroll
    for (int h = 0; h < H; ++h) d += ed[h] * scur[h];
    const float w = fmaxf(d, 0.f);
    out[(size_t)t * V + v] = w;

    if (t == L - 1) return;                  // uniform exit: last row needs no xe

    // --- partial xe: shuffle tree per wave, then LDS ds_add_f32 ---
    float p[H];
#pragma unroll
    for (int h = 0; h < H; ++h) p[h] = ee[h] * w;
#pragma unroll
    for (int off = 16; off > 0; off >>= 1)
#pragma unroll
      for (int h = 0; h < H; ++h) p[h] += __shfl_down(p[h], off);
    if (lane == 0)
#pragma unroll
      for (int h = 0; h < H; ++h) atomicAdd(&red[h], p[h]);
    __syncthreads();

    if (tid < H) part[wg * H + tid] = red[tid];
    __threadfence();
    if (tid == 0) {
      int old = __hip_atomic_fetch_add(&cnt[t], 1, __ATOMIC_ACQ_REL,
                                       __HIP_MEMORY_SCOPE_AGENT);
      if (old == NWG - 1) islast = 1;
    }
    __syncthreads();

    if (islast) {   // uniform within WG
      // sum of NWG per-WG partials (wave shuffle tree, then LDS atomicAdd)
      if (tid < H) red[tid] = 0.f;
      __syncthreads();
      float q[H] = {};
      if (tid < NWG)
#pragma unroll
        for (int h = 0; h < H; ++h) q[h] = part[tid * H + h];
#pragma unroll
      for (int off = 16; off > 0; off >>= 1)
#pragma unroll
        for (int h = 0; h < H; ++h) q[h] += __shfl_down(q[h], off);
      if (lane == 0)
#pragma unroll
        for (int h = 0; h < H; ++h) atomicAdd(&red[h], q[h]);
      __syncthreads();   // red[] = xe_t

      // GRU update (same two-phase scheme as the encoder)
      if (tid < 2 * H) {
        const int g = tid / H, i = tid % H;
        float pre = Bv[g * H + i];
#pragma unroll
        for (int j = 0; j < H; ++j)
          pre += W[(g * H + i) * H + j] * red[j] +
                 U[(g * H + i) * H + j] * scur[j];
        zr[tid] = 1.f / (1.f + __expf(-pre));
      }
      __syncthreads();
      if (tid < H) {
        float pre = Bv[2 * H + tid];
#pragma unroll
        for (int j = 0; j < H; ++j)
          pre += W[(2 * H + tid) * H + j] * red[j] +
                 U[(2 * H + tid) * H + j] * (scur[j] * zr[H + j]);
        const float c  = tanhf(pre);
        const float sn = zr[tid] * scur[tid] + (1.f - zr[tid]) * c;
        sbuf[(t + 1) * H + tid] = sn;
      }
      __threadfence();
      __syncthreads();
      if (tid == 0)
        __hip_atomic_store(&ready[t + 1], 1, __ATOMIC_RELEASE,
                           __HIP_MEMORY_SCOPE_AGENT);
    }
  }
}

// ---------------------------------------------------------------------------
// launch
// ---------------------------------------------------------------------------
extern "C" void kernel_launch(void* const* d_in, const int* in_sizes, int n_in,
                              void* d_out, int out_size, void* d_ws, size_t ws_size,
                              hipStream_t stream) {
  (void)in_sizes; (void)n_in; (void)out_size; (void)ws_size;
  const float* x     = (const float*)d_in[0];
  /* d_in[1] = l (==1024, compile-time) */
  const float* Eg_en = (const float*)d_in[2];
  const float* Eg_de = (const float*)d_in[3];
  const float* cg_de = (const float*)d_in[4];
  const float* Wg_en = (const float*)d_in[5];
  const float* Ug_en = (const float*)d_in[6];
  const float* bg_en = (const float*)d_in[7];
  const float* Wg_de = (const float*)d_in[8];
  const float* Ug_de = (const float*)d_in[9];
  const float* bg_de = (const float*)d_in[10];
  float* out = (float*)d_out;

  char* ws = (char*)d_ws;
  size_t off = 0;
  auto alloc = [&](size_t bytes) -> void* {
    void* p = ws + off;
    off = (off + bytes + 255) & ~(size_t)255;
    return p;
  };
  float* part1  = (float*)alloc(sizeof(float) * KCH * T * H);  // 820 KB
  float* xe_all = (float*)alloc(sizeof(float) * T * H);
  float* sbuf   = (float*)alloc(sizeof(float) * L * H);
  float* partd  = (float*)alloc(sizeof(float) * NWG * H);
  int*   ready  = (int*)alloc(sizeof(int) * L);
  int*   cnt    = (int*)alloc(sizeof(int) * L);

  hipLaunchKernelGGL(k0_init, dim3((L + 255) / 256), dim3(256), 0, stream,
                     ready, cnt);
  hipLaunchKernelGGL(k1_xe_wmma, dim3(TILES * KCH), dim3(32), 0, stream,
                     x, Eg_en, part1);
  hipLaunchKernelGGL(k1b_reduce, dim3((T * H + 255) / 256), dim3(256), 0, stream,
                     part1, xe_all);
  hipLaunchKernelGGL(k2_encoder, dim3(1), dim3(128), 0, stream,
                     xe_all, Wg_en, Ug_en, bg_en, sbuf, ready);
  hipLaunchKernelGGL(k3_decoder, dim3(NWG), dim3(TPB), 0, stream,
                     Eg_en, Eg_de, cg_de, Wg_de, Ug_de, bg_de,
                     out, sbuf, partd, ready, cnt);
}
